// MultiheadAttention2_68195490726015
// MI455X (gfx1250) — compile-verified, matrix operand order checked by reading, not run on hardware
//
#include <hip/hip_runtime.h>
#include <math.h>

// ---------------------------------------------------------------------------
// MultiheadAttention for MI455X (gfx1250, wave32, WMMA).
// B=8, N=1024, F=256, H=8, E=64.  Outputs: attn [H,B,N,N] fp32 then
// out [B,N,H*E] fp32, concatenated flat in d_out.
// ---------------------------------------------------------------------------

#define BB 8
#define NN 1024
#define FF 256
#define HH 8
#define EE 64
#define SCALING 0.125f  // 1/sqrt(64)

typedef unsigned short u16;
typedef __attribute__((ext_vector_type(16))) __bf16 v16bf;
typedef __attribute__((ext_vector_type(8)))  __bf16 v8bf;
typedef __attribute__((ext_vector_type(8)))  float  v8f;

// fp32 -> bf16 round-to-nearest-even (bit pattern)
static __device__ __forceinline__ u16 f2bf(float f) {
  unsigned int u = __float_as_uint(f);
  unsigned int r = u + 0x7FFFu + ((u >> 16) & 1u);
  return (u16)(r >> 16);
}

static __device__ __forceinline__ __bf16 f2bf16(float f) {
  union { u16 u; __bf16 b; } c;
  c.u = f2bf(f);
  return c.b;
}

// Two contiguous 8xbf16 (16B) runs -> one v16bf fragment.
// Slot e of v16bf covers K = (e&7) + (e>=8 ? 16 : 0) + (lane>=16 ? 8 : 0),
// i.e. slots 0..7 = [c0, c0+8), slots 8..15 = [c0+16, c0+24).
static __device__ __forceinline__ v16bf frag_from(const u16* p_c0,
                                                  const u16* p_c0p16) {
  v8bf lo = *reinterpret_cast<const v8bf*>(p_c0);
  v8bf hi = *reinterpret_cast<const v8bf*>(p_c0p16);
  v16bf r;
#pragma unroll
  for (int i = 0; i < 8; ++i) { r[i] = lo[i]; r[i + 8] = hi[i]; }
  return r;
}

// ---------------------------------------------------------------------------
// Kernel 0a: x fp32 -> bf16 (grid-stride elementwise)
// ---------------------------------------------------------------------------
__global__ void cvt_x_kernel(const float* __restrict__ x, u16* __restrict__ xbf,
                             int n) {
  for (int i = blockIdx.x * blockDim.x + threadIdx.x; i < n;
       i += gridDim.x * blockDim.x)
    xbf[i] = f2bf(x[i]);
}

// ---------------------------------------------------------------------------
// Kernel 0b: W[3][H,F,E] fp32 -> wt[3][H][E][F] bf16  (transpose so
// B-fragments load contiguously along K=F)
// ---------------------------------------------------------------------------
__global__ void cvt_w_kernel(const float* __restrict__ Wq,
                             const float* __restrict__ Wk,
                             const float* __restrict__ Wv,
                             u16* __restrict__ wt, int n) {
  for (int i = blockIdx.x * blockDim.x + threadIdx.x; i < n;
       i += gridDim.x * blockDim.x) {
    int f = i % FF;
    int e = (i / FF) % EE;
    int h = (i / (FF * EE)) % HH;
    int sel = i / (FF * EE * HH);
    const float* W = (sel == 0) ? Wq : (sel == 1) ? Wk : Wv;
    wt[i] = f2bf(W[((size_t)h * FF + f) * EE + e]);
  }
}

// ---------------------------------------------------------------------------
// Kernel 1: projections q/k/v.  One wave per (sel, h*b, 16-row tile).
//   C[16x64] = x_tile[16x256] @ W[256x64], K-chunks of 32 -> 8*4 WMMAs.
//   sel 0 -> q [h,b,n,e] bf16 (A-friendly row major)
//   sel 1 -> k [h,b,n,e] bf16 (B-friendly: lane=col of Kᵀ reads k rows)
//   sel 2 -> vt[h,b,e,n] bf16 (transposed for PV B-fragments)
// ---------------------------------------------------------------------------
__global__ __launch_bounds__(32) void proj_kernel(
    const u16* __restrict__ xbf, const u16* __restrict__ wt,
    u16* __restrict__ q, u16* __restrict__ k, u16* __restrict__ vt) {
  const int sel = blockIdx.x % 3;
  const int hb = blockIdx.x / 3;           // h*B + b
  const int row0 = blockIdx.y * 16;
  const int lane = threadIdx.x;
  const int m = lane & 15;
  const int half = lane >> 4;
  const int h = hb >> 3, b = hb & 7;

  v8f acc[4];
#pragma unroll
  for (int t = 0; t < 4; ++t) acc[t] = (v8f)(0.f);

  const u16* xrow = xbf + ((size_t)b * NN + row0 + m) * FF;
  const u16* wbase = wt + ((size_t)sel * HH + h) * EE * FF;

  for (int kc = 0; kc < 8; ++kc) {           // K = 256 in chunks of 32
    const int c0 = kc * 32 + half * 8;
    v16bf a = frag_from(xrow + c0, xrow + c0 + 16);
#pragma unroll
    for (int t = 0; t < 4; ++t) {
      const u16* wrow = wbase + (size_t)(t * 16 + m) * FF;  // Bᵀ row = out col
      v16bf bfr = frag_from(wrow + c0, wrow + c0 + 16);
      acc[t] = __builtin_amdgcn_wmma_f32_16x16x32_bf16(
          false, a, false, bfr, (short)0, acc[t], false, false);
    }
  }

  if (sel < 2) {
    u16* dst = (sel == 0 ? q : k) + ((size_t)hb * NN + row0) * EE;
#pragma unroll
    for (int t = 0; t < 4; ++t)
#pragma unroll
      for (int r = 0; r < 8; ++r)
        dst[(size_t)(r + half * 8) * EE + t * 16 + m] = f2bf(acc[t][r]);
  } else {
    // transposed store: element (M, n) -> vt[n][row0+M]; 8 consecutive bf16
    u16* vbase = vt + (size_t)hb * EE * NN;
#pragma unroll
    for (int t = 0; t < 4; ++t) {
      u16 tmp[8];
#pragma unroll
      for (int r = 0; r < 8; ++r) tmp[r] = f2bf(acc[t][r]);
      *reinterpret_cast<uint4*>(vbase + (size_t)(t * 16 + m) * NN + row0 +
                                half * 8) = *reinterpret_cast<uint4*>(tmp);
    }
  }
}

// ---------------------------------------------------------------------------
// Kernel 2: attention.  128 threads (4 waves) cooperate on one 16-row tile:
//   QK^T : wave w computes key-column tiles j = 16w .. 16w+15  (2 WMMAs each)
//          -> fp32 scores S[16][1024] in LDS
//   softmax: 128-thread two-pass reduction (8 segments/row via LDS partials)
//   attn writeout: coalesced, 128 elems/thread
//   PV   : wave w owns output col tile t=w over full K=1024 (32 WMMAs);
//          A-fragments built from LDS S with rinv[m]*SCALING folded into the
//          f32->bf16 conversion (no separate P buffer -> ~65KB LDS ->
//          4 WGs x 4 waves = 16 waves/WGP for latency hiding)
// ---------------------------------------------------------------------------
__global__ __launch_bounds__(128) void attn_kernel(
    const u16* __restrict__ q, const u16* __restrict__ k,
    const u16* __restrict__ vt, float* __restrict__ attn_out,
    float* __restrict__ out) {
  extern __shared__ __align__(16) char smem[];
  float* S = reinterpret_cast<float*>(smem);                       // 64 KiB
  float* red = reinterpret_cast<float*>(smem + 16 * NN * 4);       // 2x128 f32
  float* rinv = red + 256;                                         // 16 f32

  const int hb = blockIdx.x;               // h*B + b
  const int h = hb >> 3, b = hb & 7;
  const int row0 = blockIdx.y * 16;
  const int tid = threadIdx.x;
  const int w = tid >> 5;                  // wave id 0..3
  const int lane = tid & 31;
  const int m = lane & 15;
  const int half = lane >> 4;

  // ---- A fragments for the tile's 16 q rows (E=64 -> 2 chunks of 32) ----
  const u16* qrow = q + ((size_t)hb * NN + row0 + m) * EE;
  v16bf aq0 = frag_from(qrow + half * 8, qrow + half * 8 + 16);
  v16bf aq1 = frag_from(qrow + 32 + half * 8, qrow + 32 + half * 8 + 16);

  // ---- scores: wave w fills S[:, 256w .. 256w+255] ----
  const u16* kmat = k + (size_t)hb * NN * EE;
#pragma unroll 2
  for (int jj = 0; jj < 16; ++jj) {
    const int j = w * 16 + jj;
    const u16* krow = kmat + (size_t)(j * 16 + m) * EE;  // Bᵀ row = key row
    if (jj + 1 < 16)
      __builtin_prefetch(kmat + (size_t)((j + 1) * 16 + m) * EE, 0, 0);
    v8f c = (v8f)(0.f);
    v16bf b0 = frag_from(krow + half * 8, krow + half * 8 + 16);
    c = __builtin_amdgcn_wmma_f32_16x16x32_bf16(false, aq0, false, b0,
                                                (short)0, c, false, false);
    v16bf b1 = frag_from(krow + 32 + half * 8, krow + 32 + half * 8 + 16);
    c = __builtin_amdgcn_wmma_f32_16x16x32_bf16(false, aq1, false, b1,
                                                (short)0, c, false, false);
#pragma unroll
    for (int r = 0; r < 8; ++r)
      S[(size_t)(r + half * 8) * NN + j * 16 + m] = c[r];
  }
  __syncthreads();

  // ---- softmax over the 16 rows: 8 threads per row, 128 cols each ----
  {
    const int row = tid & 15, seg = tid >> 4;
    float* srow = S + (size_t)row * NN + seg * 128;
    float mx = -3.402823466e+38f;
    for (int c = 0; c < 128; ++c) mx = fmaxf(mx, srow[c]);
    red[seg * 16 + row] = mx;
    __syncthreads();
    float rmax = red[row];
#pragma unroll
    for (int s2 = 1; s2 < 8; ++s2) rmax = fmaxf(rmax, red[s2 * 16 + row]);
    float sm = 0.f;
    for (int c = 0; c < 128; ++c) {
      float e = __expf(srow[c] - rmax);
      srow[c] = e;
      sm += e;
    }
    red[128 + seg * 16 + row] = sm;
    __syncthreads();
    if (seg == 0) {
      float t = 0.f;
#pragma unroll
      for (int s2 = 0; s2 < 8; ++s2) t += red[128 + s2 * 16 + row];
      rinv[row] = 1.0f / t;
    }
  }
  __syncthreads();

  // ---- coalesced attn writeout: p = exp * rinv[row] ----
  {
    float* arow = attn_out + ((size_t)hb * NN + row0) * NN;
    for (int i = 0; i < 128; ++i) {
      int idx = i * 128 + tid;
      int row = idx >> 10, col = idx & 1023;
      arow[(size_t)row * NN + col] = S[idx] * rinv[row];
    }
  }
  __syncthreads();

  // ---- O = P @ V : wave w owns output cols [16w, 16w+16), K=1024 ----
  {
    const float scale = rinv[m] * SCALING;  // A row m is per-lane constant
    const u16* vrow = vt + ((size_t)hb * EE + w * 16 + m) * NN;  // Bᵀ row
    const float* prow = S + (size_t)m * NN;
    v8f o = (v8f)(0.f);
    for (int kc = 0; kc < 32; ++kc) {
      const int c0 = kc * 32 + half * 8;
      // A-fragment: 16 fp32 from LDS, scaled + converted to bf16
      float4 f0 = *reinterpret_cast<const float4*>(prow + c0);
      float4 f1 = *reinterpret_cast<const float4*>(prow + c0 + 4);
      float4 f2 = *reinterpret_cast<const float4*>(prow + c0 + 16);
      float4 f3 = *reinterpret_cast<const float4*>(prow + c0 + 20);
      v16bf a;
      a[0] = f2bf16(f0.x * scale);  a[1] = f2bf16(f0.y * scale);
      a[2] = f2bf16(f0.z * scale);  a[3] = f2bf16(f0.w * scale);
      a[4] = f2bf16(f1.x * scale);  a[5] = f2bf16(f1.y * scale);
      a[6] = f2bf16(f1.z * scale);  a[7] = f2bf16(f1.w * scale);
      a[8] = f2bf16(f2.x * scale);  a[9] = f2bf16(f2.y * scale);
      a[10] = f2bf16(f2.z * scale); a[11] = f2bf16(f2.w * scale);
      a[12] = f2bf16(f3.x * scale); a[13] = f2bf16(f3.y * scale);
      a[14] = f2bf16(f3.z * scale); a[15] = f2bf16(f3.w * scale);
      v16bf bfr = frag_from(vrow + c0, vrow + c0 + 16);
      o = __builtin_amdgcn_wmma_f32_16x16x32_bf16(false, a, false, bfr,
                                                  (short)0, o, false, false);
    }
    // out[b][n][h][e], n = row0 + M, e = 16w + m
#pragma unroll
    for (int r = 0; r < 8; ++r)
      out[(((size_t)b * NN + row0 + r + half * 8) * HH + h) * EE + w * 16 + m] =
          o[r];
  }
}

// ---------------------------------------------------------------------------
extern "C" void kernel_launch(void* const* d_in, const int* in_sizes, int n_in,
                              void* d_out, int out_size, void* d_ws,
                              size_t ws_size, hipStream_t stream) {
  const float* x = (const float*)d_in[0];
  const float* Wq = (const float*)d_in[1];
  const float* Wk = (const float*)d_in[2];
  const float* Wv = (const float*)d_in[3];

  // workspace layout (bytes)
  char* ws = (char*)d_ws;
  u16* xbf = (u16*)(ws);                               // 4 MiB
  u16* wt = (u16*)(ws + (size_t)4 * 1024 * 1024);      // 768 KiB
  u16* q = (u16*)(ws + 4980736);                       // 8 MiB
  u16* k = (u16*)(ws + 13369344);                      // 8 MiB
  u16* vt = (u16*)(ws + 21757952);                     // 8 MiB

  float* attn_out = (float*)d_out;                              // H*B*N*N
  float* out = attn_out + (size_t)HH * BB * NN * NN;            // B*N*H*E

  const int nx = BB * NN * FF;
  cvt_x_kernel<<<512, 256, 0, stream>>>(x, xbf, nx);

  const int nw = 3 * HH * EE * FF;
  cvt_w_kernel<<<384, 256, 0, stream>>>(Wq, Wk, Wv, wt, nw);

  dim3 pgrid(3 * HH * BB, NN / 16);      // 192 x 64 waves
  proj_kernel<<<pgrid, 32, 0, stream>>>(xbf, wt, q, k, vt);

  dim3 agrid(HH * BB, NN / 16);          // 64 x 64 blocks of 4 waves
  const size_t lds = (size_t)16 * NN * 4 + 256 * 4 + 16 * 4;  // ~65.1 KiB
  attn_kernel<<<agrid, 128, lds, stream>>>(q, k, vt, attn_out, out);
}
